// XConv_45801531244648
// MI455X (gfx1250) — matrix operand order repaired and checked by reading, not compiled, not run on hardware
//
#include <hip/hip_runtime.h>
#include <hip/hip_bf16.h>

// ---- problem constants (match reference) ----
#define Bq    8
#define Nq    8192
#define Mq    2048
#define Kq    16
#define CDq   64
#define CFq   64
#define CINq  128     // CD + CF
#define OUTq  128
#define DMq   4

typedef __attribute__((ext_vector_type(16))) _Float16 v16h;
typedef __attribute__((ext_vector_type(8)))  _Float16 v8h;
typedef __attribute__((ext_vector_type(8)))  float    v8f;

__device__ __forceinline__ float elu1(float v) {
    return v > 0.0f ? v : (__expf(v) - 1.0f);
}

// ---------------------------------------------------------------------------
// Kernel 0: f32 -> f16 weight conversion (one-shot, tiny)
// ---------------------------------------------------------------------------
__global__ void cvt_f16_kernel(const float* __restrict__ src, _Float16* __restrict__ dst, int n) {
    int i = blockIdx.x * 256 + threadIdx.x;
    if (i < n) dst[i] = (_Float16)src[i];
}

// ---------------------------------------------------------------------------
// Kernel 1: brute-force KNN. One thread per center, LDS-tiled candidate scan,
// register-resident top-32 insertion list (fully unrolled -> stays in VGPRs).
// idx[b][m][k] = sorted_by_dist[2*k]  (DIL=2 dilation)
// ---------------------------------------------------------------------------
__global__ void knn_kernel(const float* __restrict__ cc, const float* __restrict__ pc,
                           int* __restrict__ idx) {
    __shared__ float sx[256], sy[256], sz[256];
    const int blocksPerB = Mq / 256;                 // 8
    const int b = blockIdx.x / blocksPerB;
    const int m = (blockIdx.x % blocksPerB) * 256 + threadIdx.x;
    const float* pcb = pc + (size_t)b * 3 * Nq;

    const float cx = cc[((size_t)b * 3 + 0) * Mq + m];
    const float cy = cc[((size_t)b * 3 + 1) * Mq + m];
    const float cz = cc[((size_t)b * 3 + 2) * Mq + m];
    const float c2 = cx * cx + cy * cy + cz * cz;

    float bd[32];
    int   bi[32];
#pragma unroll
    for (int j = 0; j < 32; ++j) { bd[j] = 3.0e38f; bi[j] = 0; }

    for (int n0 = 0; n0 < Nq; n0 += 256) {
        __syncthreads();
        sx[threadIdx.x] = pcb[0 * Nq + n0 + threadIdx.x];
        sy[threadIdx.x] = pcb[1 * Nq + n0 + threadIdx.x];
        sz[threadIdx.x] = pcb[2 * Nq + n0 + threadIdx.x];
        __syncthreads();
        for (int t = 0; t < 256; ++t) {
            const float px = sx[t], py = sy[t], pz = sz[t];
            const float p2 = px * px + py * py + pz * pz;
            const float d  = c2 - 2.0f * (cx * px + cy * py + cz * pz) + p2;
            if (d < bd[31]) {
                float vd = d; int vi = n0 + t;
#pragma unroll
                for (int j = 0; j < 32; ++j) {
                    if (vd < bd[j]) {
                        float td = bd[j]; int ti = bi[j];
                        bd[j] = vd; bi[j] = vi;
                        vd = td; vi = ti;
                    }
                }
            }
        }
    }
#pragma unroll
    for (int k = 0; k < Kq; ++k)
        idx[((size_t)b * Mq + m) * Kq + k] = bi[2 * k];
}

// ---------------------------------------------------------------------------
// Kernel 2: gather + coordinate MLP (d1,d2) + feature gather -> fa (f16)
// fa layout (m-major): fa[b][m][c*K + j]  -> WMMA B fragments become two
// contiguous 128-bit loads per lane.
// ---------------------------------------------------------------------------
__global__ void fa_kernel(const float* __restrict__ cc, const float* __restrict__ pc,
                          const float* __restrict__ feat,
                          const float* __restrict__ w_d1, const float* __restrict__ b_d1,
                          const float* __restrict__ w_d2, const float* __restrict__ b_d2,
                          const int* __restrict__ idx, _Float16* __restrict__ fa) {
    __shared__ float sw1[CDq * 3], sb1[CDq], sw2[CDq * CDq], sb2[CDq];
    for (int i = threadIdx.x; i < CDq * 3; i += 256) sw1[i] = w_d1[i];
    for (int i = threadIdx.x; i < CDq; i += 256) { sb1[i] = b_d1[i]; sb2[i] = b_d2[i]; }
    for (int i = threadIdx.x; i < CDq * CDq; i += 256) sw2[i] = w_d2[i];
    __syncthreads();

    const int tid = blockIdx.x * 256 + threadIdx.x;      // == ((b*M+m)*K + k)
    const int b = tid >> 15;                             // M*K = 32768
    const int rem = tid & 32767;
    const int m = rem >> 4, k = rem & 15;
    const int i = idx[tid];

    const float l0 = pc[((size_t)b * 3 + 0) * Nq + i] - cc[((size_t)b * 3 + 0) * Mq + m];
    const float l1 = pc[((size_t)b * 3 + 1) * Nq + i] - cc[((size_t)b * 3 + 1) * Mq + m];
    const float l2 = pc[((size_t)b * 3 + 2) * Nq + i] - cc[((size_t)b * 3 + 2) * Mq + m];

    float h[CDq];
#pragma unroll
    for (int c = 0; c < CDq; ++c)
        h[c] = elu1(sw1[c * 3 + 0] * l0 + sw1[c * 3 + 1] * l1 + sw1[c * 3 + 2] * l2 + sb1[c]);

    _Float16* farow = fa + ((size_t)b * Mq + m) * (CINq * Kq);   // m-major row
    for (int o = 0; o < CDq; ++o) {
        float v = sb2[o];
#pragma unroll
        for (int c = 0; c < CDq; ++c) v += sw2[o * CDq + c] * h[c];
        farow[o * Kq + k] = (_Float16)elu1(v);
    }
    const float* fb = feat + (size_t)b * CFq * Nq;
    for (int cf = 0; cf < CFq; ++cf)
        farow[(CDq + cf) * Kq + k] = (_Float16)fb[(size_t)cf * Nq + i];
}

// ---------------------------------------------------------------------------
// WMMA GEMM:  out[b][o][m] = act( sum_k W[o][k] * X[b][m][k] + bias[o] )
// Wh: f16 OxKd row-major.  Xt: f16 MxKd row-major per batch (m-major).
// One wave computes a 16(o) x 64(m) strip, software-pipelined:
// iteration k issues the 10 b128 loads for step k+32 *before* the four
// v_wmma_f32_16x16x32_f16 that consume step k's fragments, so the WMMA chain
// covers the VMEM latency and the waits need not drain to zero.
// Prefetches are unconditional (speculative prefetch drops bad translations).
// ---------------------------------------------------------------------------
#define GWAVES 4

__device__ __forceinline__ v16h frag_from(const _Float16* p, int gap) {
    v8h lo = *(const v8h*)(p);
    v8h hi = *(const v8h*)(p + gap);
    return __builtin_shufflevector(lo, hi, 0,1,2,3,4,5,6,7,8,9,10,11,12,13,14,15);
}

__global__ void gemm_wmma_f16(const _Float16* __restrict__ Wh, const _Float16* __restrict__ Xt,
                              const float* __restrict__ bias, float* __restrict__ out,
                              int O, int Kd, int Md,
                              int batchStrideX, int batchStrideOut, int doElu) {
    const int wave  = threadIdx.x >> 5;
    const int lane  = threadIdx.x & 31;
    const int b     = blockIdx.z;
    const int oTile = (blockIdx.y * GWAVES + wave) * 16;
    const int mTile = blockIdx.x * 64;
    if (oTile >= O) return;

    const _Float16* Xb   = Xt  + (size_t)b * batchStrideX;
    float*          outb = out + (size_t)b * batchStrideOut;

    v8f acc[4] = {};
    const int row  = lane & 15;
    const int aSel = (lane < 16) ? 0 : 8;    // A-matrix K offset for high lanes
    const int bSel = (lane < 16) ? 0 : 8;    // B-matrix K offset for high lanes

    // A lane: k = k0+aSel+0..7 and k0+16+aSel+0..7 (gap 16)
    const _Float16* wrowBase = Wh + (size_t)(oTile + row) * Kd + aSel;
    // B lane: column m = mTile + t*16 + (lane&15), k = k0+bSel+0..7 and k0+16+bSel+0..7 (gap 16)
    const _Float16* xrow[4];
#pragma unroll
    for (int t = 0; t < 4; ++t)
        xrow[t] = Xb + (size_t)(mTile + t * 16 + (lane & 15)) * Kd + bSel;

    // ---- prologue: fragments for k0 = 0 ----
    v16h aCur = frag_from(wrowBase, 16);
    v16h bCur[4];
#pragma unroll
    for (int t = 0; t < 4; ++t) bCur[t] = frag_from(xrow[t], 16);

    // ---- pipelined main loop ----
    int k0 = 0;
    for (; k0 + 32 < Kd; k0 += 32) {
        const int kn = k0 + 32;
        v16h aNxt = frag_from(wrowBase + kn, 16);
        v16h bNxt[4];
#pragma unroll
        for (int t = 0; t < 4; ++t) {
            bNxt[t] = frag_from(xrow[t] + kn, 16);
            __builtin_prefetch(xrow[t] + kn + 32, 0, 1);   // speculative, unguarded
        }
#pragma unroll
        for (int t = 0; t < 4; ++t)
            acc[t] = __builtin_amdgcn_wmma_f32_16x16x32_f16(
                false, aCur, false, bCur[t], (short)0, acc[t], false, false);
        aCur = aNxt;
#pragma unroll
        for (int t = 0; t < 4; ++t) bCur[t] = bNxt[t];
    }
    // ---- epilogue: last k-step ----
#pragma unroll
    for (int t = 0; t < 4; ++t)
        acc[t] = __builtin_amdgcn_wmma_f32_16x16x32_f16(
            false, aCur, false, bCur[t], (short)0, acc[t], false, false);

    // D layout: VGPR i, lane l -> col = l&15, rowInTile = i + (l<16 ? 0 : 8)
    const int colD   = lane & 15;
    const int rowOff = (lane < 16) ? 0 : 8;
#pragma unroll
    for (int t = 0; t < 4; ++t) {
#pragma unroll
        for (int i = 0; i < 8; ++i) {
            const int o = oTile + rowOff + i;
            float v = acc[t][i] + bias[o];
            if (doElu) v = elu1(v);
            outb[(size_t)o * Md + mTile + t * 16 + colD] = v;
        }
    }
}

// ---------------------------------------------------------------------------
// Kernel 3: per-point T-net refinement (w_t2 then w_t3).
// Both stages stay within one c-slice, so one thread per (b,m,c) does two
// 16x16 matvecs with LDS-resident weights.
// Output: trans[b][m][k][l]  (k = c, l = q2), f32.
// ---------------------------------------------------------------------------
__global__ void tnet_kernel(const float* __restrict__ t_in,
                            const float* __restrict__ w_t2, const float* __restrict__ b_t2,
                            const float* __restrict__ w_t3, const float* __restrict__ b_t3,
                            float* __restrict__ trans) {
    __shared__ float sw2[4096], sw3[4096], sb2[256], sb3[256];
    for (int i = threadIdx.x; i < 4096; i += 256) { sw2[i] = w_t2[i]; sw3[i] = w_t3[i]; }
    if (threadIdx.x < 256) { sb2[threadIdx.x] = b_t2[threadIdx.x]; sb3[threadIdx.x] = b_t3[threadIdx.x]; }
    __syncthreads();

    const int tid = blockIdx.x * 256 + threadIdx.x;      // over B*M*16
    const int b = tid >> 15;
    const int rem = tid & 32767;
    const int m = rem >> 4, c = rem & 15;

    const float* tb = t_in + (size_t)b * 256 * Mq;
    float x[16];
#pragma unroll
    for (int j = 0; j < 16; ++j) x[j] = tb[(size_t)(c * 16 + j) * Mq + m];

    float s1[16];
#pragma unroll
    for (int q = 0; q < 16; ++q) {
        float v = sb2[c * 16 + q];
#pragma unroll
        for (int j = 0; j < 16; ++j) v += sw2[(c * 16 + q) * 16 + j] * x[j];
        s1[q] = elu1(v);
    }
    float* trb = trans + (((size_t)b * Mq + m) * 16 + c) * 16;
#pragma unroll
    for (int q = 0; q < 16; ++q) {
        float v = sb3[c * 16 + q];
#pragma unroll
        for (int j = 0; j < 16; ++j) v += sw3[(c * 16 + q) * 16 + j] * s1[j];
        trb[q] = v;                                      // no ELU after t3
    }
}

// ---------------------------------------------------------------------------
// Kernel 4: fused  fx = fa @ trans^T  and depthwise (w_dw) -> dw (f16),
// m-major layout [b][m][c*DM+d] for the pointwise WMMA GEMM.
// One block (128 threads) per (b,m); trans tile + w_dw staged in LDS.
// ---------------------------------------------------------------------------
__global__ void dw_kernel(const _Float16* __restrict__ fa, const float* __restrict__ trans,
                          const float* __restrict__ w_dw, const float* __restrict__ b_dw,
                          _Float16* __restrict__ dwh) {
    __shared__ float st[256];
    __shared__ float swd[CINq * DMq * Kq];   // 8192 f32 = 32 KB
    __shared__ float sbd[CINq * DMq];
    const int bm = blockIdx.x;
    for (int i = threadIdx.x; i < CINq * DMq * Kq; i += 128) swd[i] = w_dw[i];
    for (int i = threadIdx.x; i < CINq * DMq; i += 128) sbd[i] = b_dw[i];
    {
        const float* trb = trans + (size_t)bm * 256;
        for (int i = threadIdx.x; i < 256; i += 128) st[i] = trb[i];
    }
    __syncthreads();

    const int c = threadIdx.x;               // 0..127
    const _Float16* farow = fa + (size_t)bm * (CINq * Kq) + c * Kq;  // contiguous 16
    float f[16];
#pragma unroll
    for (int l = 0; l < 16; ++l) f[l] = (float)farow[l];

    float fx[16];
#pragma unroll
    for (int k = 0; k < 16; ++k) {
        float v = 0.0f;
#pragma unroll
        for (int l = 0; l < 16; ++l) v += st[k * 16 + l] * f[l];
        fx[k] = v;
    }
    _Float16* dwrow = dwh + (size_t)bm * (CINq * DMq) + c * DMq;     // contiguous 4
#pragma unroll
    for (int d = 0; d < DMq; ++d) {
        float v = sbd[c * DMq + d];
#pragma unroll
        for (int j = 0; j < 16; ++j) v += swd[(c * DMq + d) * 16 + j] * fx[j];
        dwrow[d] = (_Float16)v;
    }
}

// ---------------------------------------------------------------------------
extern "C" void kernel_launch(void* const* d_in, const int* in_sizes, int n_in,
                              void* d_out, int out_size, void* d_ws, size_t ws_size,
                              hipStream_t stream) {
    (void)in_sizes; (void)n_in; (void)out_size; (void)ws_size;
    const float* cc   = (const float*)d_in[0];
    const float* pc   = (const float*)d_in[1];
    const float* feat = (const float*)d_in[2];
    const float* w_d1 = (const float*)d_in[3];
    const float* b_d1 = (const float*)d_in[4];
    const float* w_d2 = (const float*)d_in[5];
    const float* b_d2 = (const float*)d_in[6];
    const float* w_t1 = (const float*)d_in[7];
    const float* b_t1 = (const float*)d_in[8];
    const float* w_t2 = (const float*)d_in[9];
    const float* b_t2 = (const float*)d_in[10];
    const float* w_t3 = (const float*)d_in[11];
    const float* b_t3 = (const float*)d_in[12];
    const float* w_dw = (const float*)d_in[13];
    const float* b_dw = (const float*)d_in[14];
    const float* w_pw = (const float*)d_in[15];
    const float* b_pw = (const float*)d_in[16];
    float* out = (float*)d_out;

    char* ws = (char*)d_ws;
    size_t off = 0;
    auto take = [&](size_t bytes) -> char* {
        char* p = ws + off;
        off = (off + bytes + 255) & ~(size_t)255;
        return p;
    };
    int*      idx   = (int*)     take((size_t)Bq * Mq * Kq * sizeof(int));
    _Float16* fa    = (_Float16*)take((size_t)Bq * Mq * CINq * Kq * sizeof(_Float16));
    float*    t_buf = (float*)   take((size_t)Bq * 256 * Mq * sizeof(float));
    float*    trans = (float*)   take((size_t)Bq * Mq * 256 * sizeof(float));
    _Float16* dwh   = (_Float16*)take((size_t)Bq * Mq * CINq * DMq * sizeof(_Float16));
    _Float16* w_t1h = (_Float16*)take((size_t)256 * CINq * Kq * sizeof(_Float16));
    _Float16* w_pwh = (_Float16*)take((size_t)OUTq * CINq * DMq * sizeof(_Float16));

    // 0) weight conversion to f16 (tiny)
    cvt_f16_kernel<<<(256 * CINq * Kq + 255) / 256, 256, 0, stream>>>(w_t1, w_t1h, 256 * CINq * Kq);
    cvt_f16_kernel<<<(OUTq * CINq * DMq + 255) / 256, 256, 0, stream>>>(w_pw, w_pwh, OUTq * CINq * DMq);

    // 1) KNN indices
    knn_kernel<<<Bq * (Mq / 256), 256, 0, stream>>>(cc, pc, idx);

    // 2) coordinate MLP + feature gather -> fa (f16, m-major)
    fa_kernel<<<(Bq * Mq * Kq) / 256, 256, 0, stream>>>(
        cc, pc, feat, w_d1, b_d1, w_d2, b_d2, idx, fa);

    // 3) big GEMM: t = elu(w_t1 (256x2048) @ fa^T + b_t1)  [WMMA]
    gemm_wmma_f16<<<dim3(Mq / 64, 256 / (16 * GWAVES), Bq), 32 * GWAVES, 0, stream>>>(
        w_t1h, fa, b_t1, t_buf,
        256, CINq * Kq, Mq, Mq * CINq * Kq, 256 * Mq, 1);

    // 4) per-point T-net (w_t2, w_t3) -> trans
    tnet_kernel<<<(Bq * Mq * 16) / 256, 256, 0, stream>>>(
        t_buf, w_t2, b_t2, w_t3, b_t3, trans);

    // 5) fused fx + depthwise -> dw (f16, m-major)
    dw_kernel<<<Bq * Mq, 128, 0, stream>>>(fa, trans, w_dw, b_dw, dwh);

    // 6) pointwise GEMM: out = elu(w_pw (128x512) @ dw^T + b_pw)  [WMMA]
    gemm_wmma_f16<<<dim3(Mq / 64, OUTq / (16 * GWAVES), Bq), 32 * GWAVES, 0, stream>>>(
        w_pwh, dwh, b_pw, out,
        OUTq, CINq * DMq, Mq, Mq * CINq * DMq, OUTq * Mq, 1);
}
